// MSDeformAttn_Fusion_69355131896291
// MI455X (gfx1250) — compile-verified, matrix operand order checked by reading, not run on hardware
//
#include <hip/hip_runtime.h>
#include <hip/hip_bf16.h>
#include <stdint.h>

// MSDeformAttn fusion for MI455X (gfx1250, wave32).
//   Prepass: fp32 -> f16 pixel-major activations, f16 [co][tap][ci] weights
//   Stage 1: conv3x3 implicit GEMM, v_wmma_f32_16x16x32_f16, 2Ax2B register
//            blocking, LDS staged via GLOBAL_LOAD_ASYNC_TO_LDS_B128 (ASYNCcnt)
//   Stage 2: softmax + multi-scale bilinear deform sampling over f16 value
//            (v_fma_mix_f32 consumes f16 directly; fp32 accumulate)
//   Stage 3: conv3x3 output projection -> d_out (NCHW fp32)
// Shapes (fixed): BS=4, L=3, C=256, Hd=8, P=4, H=W=64, HW=4096.

typedef __attribute__((ext_vector_type(16))) _Float16 v16h;
typedef __attribute__((ext_vector_type(8)))  _Float16 v8h;
typedef __attribute__((ext_vector_type(8)))  float    v8f;

#define HWPIX 4096

// Per-lane async copy of 16 contiguous bytes global -> LDS (ASYNCcnt-tracked).
// GVS addressing: mem = SGPR64 base + VGPR32 byte offset.
#define ASYNC_CP_B128(ldsOff, gOff, base64)                                  \
    asm volatile("global_load_async_to_lds_b128 %0, %1, %2"                  \
                 :: "v"(ldsOff), "v"(gOff), "s"(base64) : "memory")
#define ASYNC_WAIT0()                                                        \
    asm volatile("s_wait_asynccnt 0x0" ::: "memory")

// ---------------------------------------------------------------------------
// Prepass: NCHW fp32 -> pixel-major f16  out[img][pix][ci]
// ---------------------------------------------------------------------------
__global__ __launch_bounds__(256)
void cvt_pixmajor_f16(const float* __restrict__ in, _Float16* __restrict__ out,
                      int Cin, int total)
{
    int t = blockIdx.x * 256 + threadIdx.x;
    if (t >= total) return;
    int pix  = t & 4095;
    int rest = t >> 12;
    int ci   = rest % Cin;
    int img  = rest / Cin;
    out[((size_t)img * HWPIX + pix) * Cin + ci] = (_Float16)in[t];
}

// Prepass: (O, I, 3, 3) fp32 -> f16 out[o][tap][ci]
__global__ __launch_bounds__(256)
void cvt_weights_f16(const float* __restrict__ w, _Float16* __restrict__ out,
                     int Cin, int total)
{
    int t = blockIdx.x * 256 + threadIdx.x;
    if (t >= total) return;
    int tap  = t % 9;
    int rest = t / 9;
    int ci   = rest % Cin;
    int o    = rest / Cin;
    out[((size_t)o * 9 + tap) * Cin + ci] = (_Float16)w[t];
}

// ---------------------------------------------------------------------------
// Implicit-GEMM 3x3 SAME conv. f16 in (pixel-major) / f16 weights.
// Block = 256 threads = 8 waves; WG tile = 32 Cout x 256 px (4 image rows).
// Each wave owns 32 Cout x 32 px: 2 A-fragments x 2 B-fragments -> 4
// accumulators, 4 WMMAs per 4 fragment loads (1:1 ds_load_b128-pair : wmma).
// grid = (16 row-quads, Cout/32, images)
// outMode: 0 -> fp32 NCHW (outF), 1 -> f16 pixel-major (outH)
// ---------------------------------------------------------------------------
__global__ __launch_bounds__(256)
void conv3x3_wmma_async(const _Float16* __restrict__ in,  // [img][pix][Cin]
                        const _Float16* __restrict__ w,   // [co][tap][Cin]
                        const float*    __restrict__ bias,
                        float*          __restrict__ outF,
                        _Float16*       __restrict__ outH,
                        int Cin, int Cout, int outMode)
{
    const int y0   = blockIdx.x * 4;     // first of four output rows
    const int co0  = blockIdx.y * 32;
    const int img  = blockIdx.z;
    const int tid  = threadIdx.x;
    const int lane = tid & 31;
    const int wn   = tid >> 5;           // wave 0..7 : 32-px sub-tile

    // rows y0-1 .. y0+4, x = -1..64, 32 ch   (fragment order: ci innermost)
    __shared__ _Float16 sIn[6 * 66 * 32];   // 25344 B
    __shared__ _Float16 sW [32 * 9 * 32];   // 18432 B

    v8f acc[4] = {{}, {}, {}, {}};          // [a*2+s] : a = co half, s = px half

    const unsigned long long inBase = (unsigned long long)in;
    const unsigned long long wBase  = (unsigned long long)w;
    const unsigned imgOff = (unsigned)((size_t)img * HWPIX * Cin * 2);

    const int l15   = lane & 15;   // M (A) / N (B) index within 16
    const int khalf = lane >> 4;   // K-half select per 16-bit fragment layout
    const int rowW  = wn >> 1;           // wave's output row within quad (0..3)
    const int xbW   = (wn & 1) * 32;     // wave's x base within the row

    for (int ci0 = 0; ci0 < Cin; ci0 += 32) {
        __syncthreads();   // all consumers of previous chunk done

        // ---- async-stage input: 6 rows x 66 cols x 4 chunks = 1584 x 16B
        for (int idx = tid; idx < 1584; idx += 256) {
            int ck = idx & 3;               // 8-channel chunk
            int rc = idx >> 2;              // 0..395
            int ky = rc / 66;               // 0..5
            int xi = rc - ky * 66;          // 0..65  (gx = xi-1)
            int gy = y0 + ky - 1;
            int gx = xi - 1;
            _Float16* lp = &sIn[(ky * 66 + xi) * 32 + ck * 8];
            if ((unsigned)gy < 64u && (unsigned)gx < 64u) {
                unsigned ldsOff = (unsigned)(uintptr_t)lp;
                unsigned gOff   = imgOff
                                + (unsigned)(((gy * 64 + gx) * Cin + ci0) * 2)
                                + (unsigned)(ck * 16);
                ASYNC_CP_B128(ldsOff, gOff, inBase);
            } else {
                *(float4*)lp = make_float4(0.f, 0.f, 0.f, 0.f);  // zero pad halo
            }
        }
        // ---- async-stage weights: 32co x 9tap x 4 chunks = 1152 x 16B
        for (int idx = tid; idx < 1152; idx += 256) {
            int co  = idx / 36;
            int rem = idx - co * 36;
            int tap = rem >> 2;
            int ck  = rem & 3;
            unsigned ldsOff = (unsigned)(uintptr_t)&sW[(co * 9 + tap) * 32 + ck * 8];
            unsigned gOff   = (unsigned)((((co0 + co) * 9 + tap) * Cin + ci0) * 2)
                            + (unsigned)(ck * 16);
            ASYNC_CP_B128(ldsOff, gOff, wBase);
        }
        ASYNC_WAIT0();     // this wave's async copies have landed in LDS
        __syncthreads();   // publish to all waves

        // ---- 9 taps; 2 A frags x 2 B frags -> 4 WMMAs per tap
        #pragma unroll
        for (int tap = 0; tap < 9; ++tap) {
            const int ky = tap / 3;
            const int kx = tap - ky * 3;
            v16h A[2], B[2];
            #pragma unroll
            for (int a = 0; a < 2; ++a) {
                const _Float16* arow =
                    &sW[((a * 16 + l15) * 9 + tap) * 32 + khalf * 8];
                #pragma unroll
                for (int e = 0; e < 8; ++e) {
                    A[a][e] = arow[e]; A[a][8 + e] = arow[16 + e];
                }
            }
            #pragma unroll
            for (int s = 0; s < 2; ++s) {
                const _Float16* brow =
                    &sIn[((rowW + ky) * 66 + xbW + s * 16 + l15 + kx) * 32
                         + khalf * 8];
                #pragma unroll
                for (int e = 0; e < 8; ++e) {
                    B[s][e] = brow[e]; B[s][8 + e] = brow[16 + e];
                }
            }
            #pragma unroll
            for (int a = 0; a < 2; ++a)
                #pragma unroll
                for (int s = 0; s < 2; ++s)
                    acc[a * 2 + s] = __builtin_amdgcn_wmma_f32_16x16x32_f16(
                        false, A[a], false, B[s], (short)0, acc[a * 2 + s],
                        false, false);
        }
    }

    // ---- epilogue: C layout VGPR r -> M = r + 8*(lane>=16), N = lane&15
    const int n  = lane & 15;
    const int mh = lane >> 4;
    #pragma unroll
    for (int a = 0; a < 2; ++a) {
        #pragma unroll
        for (int s = 0; s < 2; ++s) {
            const int gpix = (y0 + rowW) * 64 + xbW + s * 16 + n;
            #pragma unroll
            for (int r = 0; r < 8; ++r) {
                const int cog = co0 + a * 16 + mh * 8 + r;
                const float v = acc[a * 2 + s][r] + bias[cog];
                if (outMode)
                    outH[((size_t)img * HWPIX + gpix) * Cout + cog] = (_Float16)v;
                else
                    outF[((size_t)img * Cout + cog) * HWPIX + gpix] = v;
            }
        }
    }
}

// ---------------------------------------------------------------------------
// Softmax over L*P=12 + multi-scale bilinear deformable sampling.
// Thread = (b, hd, pix), pix fastest. value is f16 pixel-major: each bilinear
// corner is 4 x b128 loads; f16 operands feed v_fma_mix_f32, fp32 accumulate.
// Output f16 pixel-major [b][pix][c] feeds the async-staged output conv.
// ---------------------------------------------------------------------------
__device__ __forceinline__ void corner_acc(const _Float16* __restrict__ vp,
                                           float wt, float* acc)
{
    const v8h* v8 = (const v8h*)vp;     // 16B-aligned (hd*32 halves = 64B)
    #pragma unroll
    for (int q = 0; q < 4; ++q) {
        v8h t = v8[q];
        #pragma unroll
        for (int e = 0; e < 8; ++e)
            acc[q * 8 + e] = fmaf((float)t[e], wt, acc[q * 8 + e]);
    }
}

__global__ __launch_bounds__(256)
void msdeform_kernel(const _Float16* __restrict__ value, // (12, 4096, 256) f16 pm
                     const float* __restrict__ off,      // (4, 192, 4096)
                     const float* __restrict__ attn,     // (4,  96, 4096)
                     const float* __restrict__ rp,       // (4, 4096, 3, 2)
                     _Float16* __restrict__ out)         // (4, 4096, 256) f16 pm
{
    const int t   = blockIdx.x * blockDim.x + threadIdx.x;  // 131072 total
    const int pix = t & 4095;
    const int hd  = (t >> 12) & 7;
    const int b   = t >> 15;

    float lg[12];
    float mx = -3.4e38f;
    #pragma unroll
    for (int j = 0; j < 12; ++j) {
        lg[j] = attn[((size_t)b * 96 + hd * 12 + j) * HWPIX + pix];
        mx = fmaxf(mx, lg[j]);
    }
    float s = 0.f;
    #pragma unroll
    for (int j = 0; j < 12; ++j) { lg[j] = __expf(lg[j] - mx); s += lg[j]; }
    const float inv = 1.0f / s;

    float acc[32];
    #pragma unroll
    for (int d = 0; d < 32; ++d) acc[d] = 0.f;

    #pragma unroll
    for (int l = 0; l < 3; ++l) {
        const float rpx = rp[(((size_t)b * HWPIX + pix) * 3 + l) * 2 + 0];
        const float rpy = rp[(((size_t)b * HWPIX + pix) * 3 + l) * 2 + 1];
        const _Float16* vbase =
            value + (size_t)(b * 3 + l) * HWPIX * 256 + hd * 32;
        #pragma unroll
        for (int p = 0; p < 4; ++p) {
            const int j = l * 4 + p;
            const float ox = off[((size_t)b * 192 + hd * 24 + j * 2 + 0) * HWPIX + pix];
            const float oy = off[((size_t)b * 192 + hd * 24 + j * 2 + 1) * HWPIX + pix];
            const float x = rpx * 64.f + ox - 0.5f;  // loc*W - 0.5
            const float y = rpy * 64.f + oy - 0.5f;
            const float xf = floorf(x), yf = floorf(y);
            const float lx = x - xf,   ly = y - yf;
            const int x0 = (int)xf, y0 = (int)yf;
            const float aw  = lg[j] * inv;
            const float w00 = (1.f - lx) * (1.f - ly) * aw;
            const float w10 = lx * (1.f - ly) * aw;
            const float w01 = (1.f - lx) * ly * aw;
            const float w11 = lx * ly * aw;
            if ((unsigned)x0     < 64u && (unsigned)y0     < 64u)
                corner_acc(vbase + ((size_t)y0 * 64 + x0) * 256,         w00, acc);
            if ((unsigned)(x0+1) < 64u && (unsigned)y0     < 64u)
                corner_acc(vbase + ((size_t)y0 * 64 + x0 + 1) * 256,     w10, acc);
            if ((unsigned)x0     < 64u && (unsigned)(y0+1) < 64u)
                corner_acc(vbase + ((size_t)(y0+1) * 64 + x0) * 256,     w01, acc);
            if ((unsigned)(x0+1) < 64u && (unsigned)(y0+1) < 64u)
                corner_acc(vbase + ((size_t)(y0+1) * 64 + x0 + 1) * 256, w11, acc);
        }
    }

    _Float16* op = out + ((size_t)b * HWPIX + pix) * 256 + hd * 32;
    #pragma unroll
    for (int d = 0; d < 32; ++d) op[d] = (_Float16)acc[d];
}

// ---------------------------------------------------------------------------
extern "C" void kernel_launch(void* const* d_in, const int* in_sizes, int n_in,
                              void* d_out, int out_size, void* d_ws, size_t ws_size,
                              hipStream_t stream)
{
    (void)in_sizes; (void)n_in; (void)out_size; (void)ws_size;

    const float* query   = (const float*)d_in[0];   // (4,768,64,64)
    const float* refpts  = (const float*)d_in[1];   // (4,4096,3,2)
    const float* in_flat = (const float*)d_in[2];   // 12 x (256,64,64)
    const float* w_value = (const float*)d_in[6];
    const float* b_value = (const float*)d_in[7];
    const float* w_off   = (const float*)d_in[8];
    const float* b_off   = (const float*)d_in[9];
    const float* w_attn  = (const float*)d_in[10];
    const float* b_attn  = (const float*)d_in[11];
    const float* w_out   = (const float*)d_in[12];
    const float* b_out   = (const float*)d_in[13];
    float* out = (float*)d_out;

    // workspace carve-up (256B-aligned regions), ~110 MB total (L2-resident)
    char* p = (char*)d_ws;
    auto alloc = [&](size_t bytes) {
        char* r = p; p += (bytes + 255) & ~(size_t)255; return (void*)r;
    };
    _Float16* inV16   = (_Float16*)alloc((size_t)12 * HWPIX * 256 * 2);  // 25 MB
    _Float16* q16     = (_Float16*)alloc((size_t)4  * HWPIX * 768 * 2);  // 25 MB
    _Float16* wv16    = (_Float16*)alloc((size_t)256 * 9 * 256 * 2);
    _Float16* wo16    = (_Float16*)alloc((size_t)192 * 9 * 768 * 2);
    _Float16* wa16    = (_Float16*)alloc((size_t)96  * 9 * 768 * 2);
    _Float16* wu16    = (_Float16*)alloc((size_t)256 * 9 * 256 * 2);
    _Float16* valueH  = (_Float16*)alloc((size_t)12 * HWPIX * 256 * 2);  // 25 MB
    float*    offBuf  = (float*)alloc((size_t)4  * 192 * HWPIX * 4);
    float*    attnBuf = (float*)alloc((size_t)4  *  96 * HWPIX * 4);
    _Float16* dout16  = (_Float16*)alloc((size_t)4 * HWPIX * 256 * 2);

    // ---- prepass conversions (f16, async-copy-friendly layouts)
    cvt_pixmajor_f16<<<49152, 256, 0, stream>>>(in_flat, inV16, 256, 12582912);
    cvt_pixmajor_f16<<<49152, 256, 0, stream>>>(query,   q16,   768, 12582912);
    cvt_weights_f16 <<< 2304, 256, 0, stream>>>(w_value, wv16, 256,  589824);
    cvt_weights_f16 <<< 5184, 256, 0, stream>>>(w_off,   wo16, 768, 1327104);
    cvt_weights_f16 <<< 2592, 256, 0, stream>>>(w_attn,  wa16, 768,  663552);
    cvt_weights_f16 <<< 2304, 256, 0, stream>>>(w_out,   wu16, 256,  589824);

    // ---- WMMA convs (async LDS staging, 2Ax2B register blocking)
    conv3x3_wmma_async<<<dim3(16, 8, 12), 256, 0, stream>>>(
        inV16, wv16, b_value, nullptr, valueH, 256, 256, 1);   // f16 pixel-major
    conv3x3_wmma_async<<<dim3(16, 6, 4), 256, 0, stream>>>(
        q16, wo16, b_off, offBuf, nullptr, 768, 192, 0);       // fp32 NCHW
    conv3x3_wmma_async<<<dim3(16, 3, 4), 256, 0, stream>>>(
        q16, wa16, b_attn, attnBuf, nullptr, 768, 96, 0);      // fp32 NCHW

    // ---- deform core -> f16 pixel-major for the output conv
    msdeform_kernel<<<512, 256, 0, stream>>>(valueH, offBuf, attnBuf, refpts, dout16);

    // ---- output projection -> d_out (fp32 NCHW)
    conv3x3_wmma_async<<<dim3(16, 8, 4), 256, 0, stream>>>(
        dout16, wu16, b_out, out, nullptr, 256, 256, 0);
}
